// SRNN_bfd_33646773797372
// MI455X (gfx1250) — matrix operand order repaired and checked by reading, not compiled, not run on hardware
//
#include <hip/hip_runtime.h>
#include <hip/hip_bf16.h>

// ---------------------------------------------------------------------------
// SRNN persistent kernel for MI455X (gfx1250, wave32).
// Dominant GEMM (spk_all @ W_big.T, K=4218, M=64) runs on
// v_wmma_f32_16x16x32_f16 with f32 accumulation; spikes are {0,1} (exact in
// f16), weights converted once to zero-padded f16. W_big (71MB) is L2-resident.
// Grid-wide sequencing (13-pop Gauss-Seidel x 100 steps) via an atomic
// monotonic-counter barrier in d_ws (reset each launch with hipMemsetAsync,
// graph-capture safe).
// ---------------------------------------------------------------------------

#define BSZ   64
#define TSTEPS 100
#define NN    4218
#define NPAD  4224      // NN padded to multiple of 32 (K for WMMA)
#define NBAR  200
#define KBAR  224       // 200 padded to multiple of 32
#define NHID  128
#define NBLK  128
#define NTHR  256
#define B_J0  0.04f
#define BETA  1.8f

typedef __attribute__((ext_vector_type(16))) _Float16 v16h;
typedef __attribute__((ext_vector_type(8)))  _Float16 v8h;
typedef __attribute__((ext_vector_type(8)))  float    v8f;

__device__ __constant__ int d_POP[13] = {440,934,94,93,2232,106,4,55,64,64,34,60,38};
__device__ __constant__ int d_OFF[14] = {0,440,1374,1468,1561,3793,3899,3903,3958,4022,4086,4120,4180,4218};

struct In {
    const float *x, *W_f, *b_f, *W_t, *b_t, *W_bar, *b_bar;
    const float *tauAdpBar, *tauMBar, *W_th, *W_big, *tauAdpAll, *tauMAll;
    const float *W21, *b21, *W22, *b22, *memB0, *spkB0, *mem0, *spk0;
};

struct WS {
    unsigned long long* cnt;
    _Float16 *Wbig16, *Wth16, *W21f16, *spkAll16, *spkB16;
    float *spkAll, *memAll, *bAll, *cur;
    float *spkB, *memB, *bB, *sweep;
    float *hid, *outAcc, *alphaAll, *roAll, *alphaBar, *roBar;
};

// ---------------- grid barrier (monotonic counter, agent scope) -------------
__device__ __forceinline__ void grid_sync(unsigned long long* cnt,
                                          unsigned long long target) {
    __syncthreads();
    if (threadIdx.x == 0) {
        __threadfence();                       // make prior writes visible
        atomicAdd(cnt, 1ull);
        while (__hip_atomic_load(cnt, __ATOMIC_RELAXED,
                                 __HIP_MEMORY_SCOPE_AGENT) < target) {
            __builtin_amdgcn_s_sleep(2);
        }
        __threadfence();                       // acquire others' writes
    }
    __syncthreads();
}

// ---------------- WMMA 16x16 tile, f16 in / f32 acc -------------------------
// A: row-major (M x lda) f16, rows = batch. B: row-major (N x ldb) f16 where
// B row n holds W[n][k] (so product computes A @ W.T). Layouts per ISA 7.12.2.
__device__ __forceinline__ void mm_tile_acc(const _Float16* __restrict__ A, int lda,
                                            const _Float16* __restrict__ B, int ldb,
                                            int mt, int nt, int Klen,
                                            v8f& c, int lane) {
    const int m  = mt * 16 + (lane & 15);
    const int hi = lane >> 4;
    const _Float16* arow = A + (size_t)m * lda;
    const _Float16* brow = B + (size_t)(nt * 16 + (lane & 15)) * ldb;
    for (int k0 = 0; k0 < Klen; k0 += 32) {
        v16h a, b;
        // A 16x32 f16 frag: lanes0-15 j0-7 -> K=k0..k0+7, j8-15 -> K=k0+16..+23
        //                   lanes16-31 shifted by +8 in K.
        ((v8h*)&a)[0] = *(const v8h*)(arow + k0 + 8 * hi);
        ((v8h*)&a)[1] = *(const v8h*)(arow + k0 + 16 + 8 * hi);
        // B 32x16 f16 frag: lane = column n, same K order as A.
        ((v8h*)&b)[0] = *(const v8h*)(brow + k0 + 8 * hi);
        ((v8h*)&b)[1] = *(const v8h*)(brow + k0 + 16 + 8 * hi);
        __builtin_prefetch(brow + k0 + 64, 0, 1);  // L2-resident weight stream
        c = __builtin_amdgcn_wmma_f32_16x16x32_f16(false, a, false, b,
                                                   (short)0, c, false, false);
    }
}

__device__ __forceinline__ void store_tile(float* __restrict__ C, int ldc,
                                           int mt, int nbase, int nmax,
                                           const v8f& c, int lane) {
    const int n = nbase + (lane & 15);
    if (n < nmax) {
        const int hi = lane >> 4;
        #pragma unroll
        for (int r = 0; r < 8; r++)
            C[(size_t)(mt * 16 + r + 8 * hi) * ldc + n] = c[r];
    }
}

__global__ __launch_bounds__(NTHR, 1) void srnn_persistent(In in, WS ws,
                                                           float* __restrict__ out) {
    const int  gtid  = blockIdx.x * blockDim.x + threadIdx.x;
    const int  GTOT  = gridDim.x * blockDim.x;
    const int  lane  = threadIdx.x & 31;
    const int  gwave = blockIdx.x * (blockDim.x >> 5) + (threadIdx.x >> 5);
    const int  NWAVE = gridDim.x * (blockDim.x >> 5);
    unsigned long long ep = 0;
    #define GBAR() do { ep++; grid_sync(ws.cnt, ep * (unsigned long long)gridDim.x); } while (0)

    // ---------------- init: f16 weight packs, state, per-neuron alpha/ro ----
    for (size_t e = gtid; e < (size_t)NPAD * NPAD; e += GTOT) {
        size_t n = e / NPAD, k = e % NPAD;
        ws.Wbig16[e] = (n < NN && k < NN) ? (_Float16)in.W_big[n * NN + k] : (_Float16)0.f;
    }
    for (size_t e = gtid; e < (size_t)NPAD * KBAR; e += GTOT) {
        size_t n = e / KBAR, k = e % KBAR;
        ws.Wth16[e] = (n < NN && k < NBAR) ? (_Float16)in.W_th[n * NBAR + k] : (_Float16)0.f;
    }
    // W21 columns map onto spk_all indices: ro_in = [pops0,1 (0..1374)] ++ [pop4 (1561..3793)]
    for (size_t e = gtid; e < (size_t)NHID * NPAD; e += GTOT) {
        size_t h = e / NPAD, kk = e % NPAD;
        int c = -1;
        if (kk < 1374)                  c = (int)kk;
        else if (kk >= 1561 && kk < 3793) c = (int)kk - 187;
        ws.W21f16[e] = (c >= 0) ? (_Float16)in.W21[h * 3606 + c] : (_Float16)0.f;
    }
    for (size_t e = gtid; e < (size_t)BSZ * NPAD; e += GTOT) {
        size_t b = e / NPAD, n = e % NPAD;
        if (n < NN) {
            float s0 = in.spk0[b * NN + n];
            ws.spkAll[e]   = s0;
            ws.memAll[e]   = in.mem0[b * NN + n];
            ws.bAll[e]     = B_J0;
            ws.spkAll16[e] = (_Float16)s0;
        } else {
            ws.spkAll[e] = 0.f; ws.memAll[e] = 0.f; ws.bAll[e] = 0.f;
            ws.spkAll16[e] = (_Float16)0.f;
        }
    }
    for (int e = gtid; e < BSZ * KBAR; e += GTOT) {
        int b = e / KBAR, n = e % KBAR;
        ws.spkB16[e] = (n < NBAR) ? (_Float16)in.spkB0[b * NBAR + n] : (_Float16)0.f;
    }
    for (int e = gtid; e < BSZ * NBAR; e += GTOT) {
        ws.spkB[e] = in.spkB0[e]; ws.memB[e] = in.memB0[e]; ws.bB[e] = B_J0;
    }
    for (int e = gtid; e < BSZ * 3; e += GTOT) ws.outAcc[e] = 0.f;
    for (int n = gtid; n < NPAD; n += GTOT) {
        ws.alphaAll[n] = (n < NN) ? expf(-1.0f / in.tauMAll[n])   : 0.f;
        ws.roAll[n]    = (n < NN) ? expf(-1.0f / in.tauAdpAll[n]) : 0.f;
    }
    for (int n = gtid; n < KBAR; n += GTOT) {
        ws.alphaBar[n] = (n < NBAR) ? expf(-1.0f / in.tauMBar[n])   : 0.f;
        ws.roBar[n]    = (n < NBAR) ? expf(-1.0f / in.tauAdpBar[n]) : 0.f;
    }
    GBAR();

    // ---------------- time loop --------------------------------------------
    for (int t = 0; t < TSTEPS; t++) {
        // P_sweep: 64x200, K=93 (tiny, scalar f32)
        for (int e = gtid; e < BSZ * NBAR; e += GTOT) {
            int b = e / NBAR, j = e % NBAR;
            int cch = (j < 100) ? 0 : 1;
            int jj  = (j < 100) ? j : j - 100;
            const float* Wr = (cch == 0 ? in.W_f : in.W_t) + jj * 93;
            const float* xr = in.x + (((size_t)b * 2 + cch) * TSTEPS + t) * 93;
            float acc = (cch == 0 ? in.b_f : in.b_t)[jj];
            for (int f = 0; f < 93; f++) acc += xr[f] * Wr[f];
            ws.sweep[e] = acc;
        }
        GBAR();

        // P_bar: 64x200, K=200 + adaptive-LIF update -> spk_b (f32 + f16)
        for (int e = gtid; e < BSZ * NBAR; e += GTOT) {
            int b = e / NBAR, n = e % NBAR;
            const float* sw = ws.sweep + b * NBAR;
            const float* Wr = in.W_bar + n * NBAR;
            float acc = in.b_bar[n];
            for (int k = 0; k < NBAR; k++) acc += sw[k] * Wr[k];
            float alpha = ws.alphaBar[n], ro = ws.roBar[n];
            float so = ws.spkB[e];
            float bb = ro * ws.bB[e] + (1.f - ro) * so;
            float Bv = B_J0 + BETA * bb;
            float mem = ws.memB[e] * alpha + (1.f - alpha) * acc - Bv * so;
            float sp = (mem - Bv) > 0.f ? 1.f : 0.f;
            ws.memB[e] = mem; ws.bB[e] = bb; ws.spkB[e] = sp;
            ws.spkB16[(size_t)b * KBAR + n] = (_Float16)sp;
        }
        GBAR();

        // Gauss-Seidel over 13 populations
        for (int i = 0; i < 13; i++) {
            const int o = d_OFF[i], s = d_POP[i];
            const int ntiles = (s + 15) >> 4;
            const int ttiles = 4 * ntiles;      // M-tiles (64/16) x N-tiles
            // P_mm(i): WMMA, cur[:, o:o+s] = spk_b@Wth.T + spk_all@Wbig.T
            for (int tt = gwave; tt < ttiles; tt += NWAVE) {
                int mt = tt & 3, nt = tt >> 2;
                v8f c = {};
                mm_tile_acc(ws.spkB16,   KBAR, ws.Wth16  + (size_t)o * KBAR, KBAR, mt, nt, KBAR, c, lane);
                mm_tile_acc(ws.spkAll16, NPAD, ws.Wbig16 + (size_t)o * NPAD, NPAD, mt, nt, NPAD, c, lane);
                store_tile(ws.cur + o, NPAD, mt, nt * 16, s, c, lane);
            }
            GBAR();
            // P_up(i): adaptive-LIF elementwise, commit new spikes (+h_state)
            for (int e = gtid; e < BSZ * s; e += GTOT) {
                int b = e / s, n = o + (e - (e / s) * s);
                size_t idx = (size_t)b * NPAD + n;
                float alpha = ws.alphaAll[n], ro = ws.roAll[n];
                float so = ws.spkAll[idx];
                float bb = ro * ws.bAll[idx] + (1.f - ro) * so;
                float Bv = B_J0 + BETA * bb;
                float mem = ws.memAll[idx] * alpha + (1.f - alpha) * ws.cur[idx] - Bv * so;
                float sp = (mem - Bv) > 0.f ? 1.f : 0.f;
                ws.memAll[idx] = mem; ws.bAll[idx] = bb; ws.spkAll[idx] = sp;
                ws.spkAll16[idx] = (_Float16)sp;
                out[192 + ((size_t)t * BSZ + b) * NN + n] = sp;  // h_state
            }
            GBAR();
        }

        // P_ro1: hid = spk_all_f16 @ W21full.T  (64x128, K=4224) via WMMA
        for (int tt = gwave; tt < 4 * (NHID / 16); tt += NWAVE) {
            int mt = tt & 3, nt = tt >> 2;
            v8f c = {};
            mm_tile_acc(ws.spkAll16, NPAD, ws.W21f16, NPAD, mt, nt, NPAD, c, lane);
            store_tile(ws.hid, NHID, mt, nt * 16, NHID, c, lane);
        }
        GBAR();

        // P_ro2: logits, softmax, accumulate (skip t==0 like reference)
        if (t > 0 && gtid < BSZ) {
            int b = gtid;
            float l[3];
            for (int oo = 0; oo < 3; oo++) {
                float acc = in.b22[oo];
                const float* wr = in.W22 + oo * NHID;
                for (int h = 0; h < NHID; h++)
                    acc += (ws.hid[b * NHID + h] + in.b21[h]) * wr[h];
                l[oo] = acc;
            }
            float m = fmaxf(l[0], fmaxf(l[1], l[2]));
            float e0 = expf(l[0] - m), e1 = expf(l[1] - m), e2 = expf(l[2] - m);
            float sinv = 1.f / (e0 + e1 + e2);
            ws.outAcc[b * 3 + 0] += e0 * sinv;
            ws.outAcc[b * 3 + 1] += e1 * sinv;
            ws.outAcc[b * 3 + 2] += e2 * sinv;
        }
        // barrier at top of next phase (P_sweep's GBAR) protects hid/outAcc
    }
    GBAR();
    for (int e = gtid; e < BSZ * 3; e += GTOT) out[e] = ws.outAcc[e];
    #undef GBAR
}

// ---------------------------------------------------------------------------
extern "C" void kernel_launch(void* const* d_in, const int* in_sizes, int n_in,
                              void* d_out, int out_size, void* d_ws, size_t ws_size,
                              hipStream_t stream) {
    (void)in_sizes; (void)n_in; (void)out_size;
    In in;
    in.x = (const float*)d_in[0];   in.W_f = (const float*)d_in[1];
    in.b_f = (const float*)d_in[2]; in.W_t = (const float*)d_in[3];
    in.b_t = (const float*)d_in[4]; in.W_bar = (const float*)d_in[5];
    in.b_bar = (const float*)d_in[6]; in.tauAdpBar = (const float*)d_in[7];
    in.tauMBar = (const float*)d_in[8]; in.W_th = (const float*)d_in[9];
    in.W_big = (const float*)d_in[10]; in.tauAdpAll = (const float*)d_in[11];
    in.tauMAll = (const float*)d_in[12]; in.W21 = (const float*)d_in[13];
    in.b21 = (const float*)d_in[14]; in.W22 = (const float*)d_in[15];
    in.b22 = (const float*)d_in[16]; in.memB0 = (const float*)d_in[17];
    in.spkB0 = (const float*)d_in[18]; in.mem0 = (const float*)d_in[19];
    in.spk0 = (const float*)d_in[20];

    char* base = (char*)d_ws;
    size_t off = 0;
    auto alloc = [&](size_t bytes) -> char* {
        char* r = base + off;
        off = (off + bytes + 255) & ~(size_t)255;
        return r;
    };
    WS ws;
    ws.cnt      = (unsigned long long*)alloc(256);
    ws.Wbig16   = (_Float16*)alloc((size_t)NPAD * NPAD * 2);   // ~35.7 MB
    ws.Wth16    = (_Float16*)alloc((size_t)NPAD * KBAR * 2);
    ws.W21f16   = (_Float16*)alloc((size_t)NHID * NPAD * 2);
    ws.spkAll16 = (_Float16*)alloc((size_t)BSZ * NPAD * 2);
    ws.spkB16   = (_Float16*)alloc((size_t)BSZ * KBAR * 2);
    ws.spkAll   = (float*)alloc((size_t)BSZ * NPAD * 4);
    ws.memAll   = (float*)alloc((size_t)BSZ * NPAD * 4);
    ws.bAll     = (float*)alloc((size_t)BSZ * NPAD * 4);
    ws.cur      = (float*)alloc((size_t)BSZ * NPAD * 4);
    ws.spkB     = (float*)alloc((size_t)BSZ * NBAR * 4);
    ws.memB     = (float*)alloc((size_t)BSZ * NBAR * 4);
    ws.bB       = (float*)alloc((size_t)BSZ * NBAR * 4);
    ws.sweep    = (float*)alloc((size_t)BSZ * NBAR * 4);
    ws.hid      = (float*)alloc((size_t)BSZ * NHID * 4);
    ws.outAcc   = (float*)alloc((size_t)BSZ * 3 * 4);
    ws.alphaAll = (float*)alloc((size_t)NPAD * 4);
    ws.roAll    = (float*)alloc((size_t)NPAD * 4);
    ws.alphaBar = (float*)alloc((size_t)KBAR * 4);
    ws.roBar    = (float*)alloc((size_t)KBAR * 4);
    (void)ws_size;  // needs ~45 MB

    hipMemsetAsync(ws.cnt, 0, sizeof(unsigned long long), stream);
    srnn_persistent<<<NBLK, NTHR, 0, stream>>>(in, ws, (float*)d_out);
}
